// TernaryLinear_70274254897412
// MI455X (gfx1250) — 8-bit matrix operand order repaired and checked by reading, compiled, not run on hardware
//
#include <hip/hip_runtime.h>
#include <hip/hip_bf16.h>

typedef int v8i __attribute__((ext_vector_type(8)));

#define KDIM 4096
#define NDIM 4096
#define MDIM 16384

__global__ __launch_bounds__(256) void quant_split_kernel(
    const float* __restrict__ x,
    unsigned char* __restrict__ hiP,
    unsigned char* __restrict__ loP,
    int n4) {
  int i = blockIdx.x * blockDim.x + threadIdx.x;
  if (i >= n4) return;
  float4 v = ((const float4*)x)[i];

  int q0 = (int)__builtin_rintf(v.x * 256.0f);
  int q1 = (int)__builtin_rintf(v.y * 256.0f);
  int q2 = (int)__builtin_rintf(v.z * 256.0f);
  int q3 = (int)__builtin_rintf(v.w * 256.0f);
  q0 = q0 < -32768 ? -32768 : (q0 > 32767 ? 32767 : q0);
  q1 = q1 < -32768 ? -32768 : (q1 > 32767 ? 32767 : q1);
  q2 = q2 < -32768 ? -32768 : (q2 > 32767 ? 32767 : q2);
  q3 = q3 < -32768 ? -32768 : (q3 > 32767 ? 32767 : q3);

  unsigned int hp = ((unsigned int)((q0 >> 8) & 0xFF)) |
                    ((unsigned int)((q1 >> 8) & 0xFF) << 8) |
                    ((unsigned int)((q2 >> 8) & 0xFF) << 16) |
                    ((unsigned int)((q3 >> 8) & 0xFF) << 24);
  unsigned int lp = ((unsigned int)(q0 & 0xFF)) |
                    ((unsigned int)(q1 & 0xFF) << 8) |
                    ((unsigned int)(q2 & 0xFF) << 16) |
                    ((unsigned int)(q3 & 0xFF) << 24);
  ((unsigned int*)hiP)[i] = hp;
  ((unsigned int*)loP)[i] = lp;
}

__global__ __launch_bounds__(256) void ternarize_transpose_kernel(
    const float* __restrict__ w,
    signed char* __restrict__ wT) {
  __shared__ signed char tile[32][33];
  int tn0 = blockIdx.x * 32;
  int tk0 = blockIdx.y * 32;
  int tx = threadIdx.x;
  int ty = threadIdx.y;
#pragma unroll
  for (int i = 0; i < 32; i += 8) {
    float v = w[(size_t)(tk0 + ty + i) * NDIM + (tn0 + tx)];
    signed char t = (v > 0.05f) ? 1 : ((v < -0.05f) ? -1 : 0);
    tile[ty + i][tx] = t;
  }
  __syncthreads();
#pragma unroll
  for (int i = 0; i < 32; i += 8) {
    wT[(size_t)(tn0 + ty + i) * KDIM + (tk0 + tx)] = tile[tx][ty + i];
  }
}

__device__ __forceinline__ v8i load_a8(const signed char* p, int k) {
  int2 d0 = *(const int2*)(p + k);
  int2 d1 = *(const int2*)(p + k + 16);
  int2 d2 = *(const int2*)(p + k + 32);
  int2 d3 = *(const int2*)(p + k + 48);
  v8i a;
  a[0] = d0.x; a[1] = d0.y; a[2] = d1.x; a[3] = d1.y;
  a[4] = d2.x; a[5] = d2.y; a[6] = d3.x; a[7] = d3.y;
  return a;
}

__device__ __forceinline__ v8i load_b8(const signed char* p, int k) {
  int2 b0 = *(const int2*)(p + k);
  int2 b1 = *(const int2*)(p + k + 16);
  int2 b2 = *(const int2*)(p + k + 32);
  int2 b3 = *(const int2*)(p + k + 48);
  v8i b;
  b[0] = b0.x; b[1] = b0.y; b[2] = b1.x; b[3] = b1.y;
  b[4] = b2.x; b[5] = b2.y; b[6] = b3.x; b[7] = b3.y;
  return b;
}

__device__ __forceinline__ void load_bchunk(const signed char* p, int k, v8i* b) {
#pragma unroll
  for (int t = 0; t < 4; ++t) b[t] = load_b8(p, k + t * 16 * KDIM);
}

__device__ __forceinline__ void wmma_chunk(const v8i& aHi, const v8i& aLo,
                                           const v8i* b, v8i* accHi, v8i* accLo) {
#pragma unroll
  for (int t = 0; t < 4; ++t) {
    accHi[t] = __builtin_amdgcn_wmma_i32_16x16x64_iu8(
        true, aHi, true, b[t], accHi[t], false, false);
    accLo[t] = __builtin_amdgcn_wmma_i32_16x16x64_iu8(
        false, aLo, true, b[t], accLo[t], false, false);
  }
}

__global__ __launch_bounds__(512) void ternary_gemm_iu8_kernel(
    const signed char* __restrict__ Ahi,
    const signed char* __restrict__ Alo,
    const signed char* __restrict__ Bt,
    float* __restrict__ out) {
  const int wave = (int)(threadIdx.x >> 5);
  const int lane = (int)(threadIdx.x & 31);
  const int l16  = lane & 15;
  const int half = lane >> 4;

  const int wave_m = wave & 7;
  const int wave_n = wave >> 3;
  const int m0 = blockIdx.y * 128 + wave_m * 16;
  const int n0 = blockIdx.x * 128 + wave_n * 64;

  const signed char* aHiP = Ahi + (size_t)(m0 + l16) * KDIM + half * 8;
  const signed char* aLoP = Alo + (size_t)(m0 + l16) * KDIM + half * 8;
  const signed char* bP   = Bt + (size_t)(n0 + l16) * KDIM + half * 8;

  v8i accHi[4] = {v8i{0,0,0,0,0,0,0,0}, v8i{0,0,0,0,0,0,0,0},
                  v8i{0,0,0,0,0,0,0,0}, v8i{0,0,0,0,0,0,0,0}};
  v8i accLo[4] = {v8i{0,0,0,0,0,0,0,0}, v8i{0,0,0,0,0,0,0,0},
                  v8i{0,0,0,0,0,0,0,0}, v8i{0,0,0,0,0,0,0,0}};

  v8i aHiA = load_a8(aHiP, 0);
  v8i aLoA = load_a8(aLoP, 0);
  v8i bA[4], bB[4];
  load_bchunk(bP, 0, bA);

  for (int kb = 0; kb < KDIM - 256; kb += 256) {
    __builtin_prefetch(aHiP + 512, 0, 3);
    __builtin_prefetch(aLoP + 512, 0, 3);
    __builtin_prefetch(bP + 512, 0, 3);

    v8i aHiB = load_a8(aHiP, 64);
    v8i aLoB = load_a8(aLoP, 64);
    load_bchunk(bP, 64, bB);
    wmma_chunk(aHiA, aLoA, bA, accHi, accLo);

    aHiA = load_a8(aHiP, 128);
    aLoA = load_a8(aLoP, 128);
    load_bchunk(bP, 128, bA);
    wmma_chunk(aHiB, aLoB, bB, accHi, accLo);

    aHiB = load_a8(aHiP, 192);
    aLoB = load_a8(aLoP, 192);
    load_bchunk(bP, 192, bB);
    wmma_chunk(aHiA, aLoA, bA, accHi, accLo);

    aHiA = load_a8(aHiP, 256);
    aLoA = load_a8(aLoP, 256);
    load_bchunk(bP, 256, bA);
    wmma_chunk(aHiB, aLoB, bB, accHi, accLo);

    aHiP += 256; aLoP += 256; bP += 256;
  }

  {
    v8i aHiB = load_a8(aHiP, 64);
    v8i aLoB = load_a8(aLoP, 64);
    load_bchunk(bP, 64, bB);
    wmma_chunk(aHiA, aLoA, bA, accHi, accLo);

    aHiA = load_a8(aHiP, 128);
    aLoA = load_a8(aLoP, 128);
    load_bchunk(bP, 128, bA);
    wmma_chunk(aHiB, aLoB, bB, accHi, accLo);

    aHiB = load_a8(aHiP, 192);
    aLoB = load_a8(aLoP, 192);
    load_bchunk(bP, 192, bB);
    wmma_chunk(aHiA, aLoA, bA, accHi, accLo);

    wmma_chunk(aHiB, aLoB, bB, accHi, accLo);
  }

#pragma unroll
  for (int t = 0; t < 4; ++t) {
    const unsigned cOff = (unsigned)((m0 + half * 8) * NDIM + n0 + 16 * t + l16);
#pragma unroll
    for (int r = 0; r < 8; ++r) {
      int v = (accHi[t][r] << 8) + accLo[t][r];
      v = v < -32768 ? -32768 : (v > 32767 ? 32767 : v);
      out[cOff + (unsigned)(r * NDIM)] = (float)v * 0.00390625f;
    }
  }
}

extern "C" void kernel_launch(void* const* d_in, const int* in_sizes, int n_in,
                              void* d_out, int out_size, void* d_ws, size_t ws_size,
                              hipStream_t stream) {
  const float* x = (const float*)d_in[0];
  const float* w = (const float*)d_in[1];
  float* out = (float*)d_out;

  unsigned char* Ahi = (unsigned char*)d_ws;
  unsigned char* Alo = Ahi + (size_t)MDIM * KDIM;
  signed char*   Bt  = (signed char*)(Alo + (size_t)MDIM * KDIM);

  const int n4 = (MDIM * KDIM) / 4;
  quant_split_kernel<<<n4 / 256, 256, 0, stream>>>(x, Ahi, Alo, n4);

  ternarize_transpose_kernel<<<dim3(NDIM / 32, KDIM / 32), dim3(32, 8), 0, stream>>>(w, Bt);

  ternary_gemm_iu8_kernel<<<dim3(NDIM / 128, MDIM / 128), 512, 0, stream>>>(
      (const signed char*)Ahi, (const signed char*)Alo, Bt, out);
}
